// GIN_41987600286250
// MI455X (gfx1250) — matrix operand order repaired and checked by reading, not compiled, not run on hardware
//
#include <hip/hip_runtime.h>
#include <hip/hip_bf16.h>

// ---------------------------------------------------------------------------
// GCN (2x GCNConv + 2x Linear) for MI455X / gfx1250.
//   - GEMMs:   v_wmma_f32_16x16x32_bf16, f32 accumulate, bf16 A/B converted
//              on the fly. Whole 16xK A tile staged in LDS once (single
//              barrier), padded stride (K+4) for conflict-free ds_load_b128;
//              K-loop is barrier-free so the scheduler can pipeline the
//              L2-resident Wt loads against the WMMAs.
//   - Graph:   gather/scale/scatter with global_atomic_add_f32; both the
//              message source and aggregation buffers (51.2 MB each) are
//              L2-resident (192 MB L2), so random traffic never hits HBM.
// ---------------------------------------------------------------------------

typedef __attribute__((ext_vector_type(16))) __bf16 v16bf;
typedef __attribute__((ext_vector_type(8)))  float  v8f;

#define NHID 128

// ---------------- small utility kernels ----------------

__global__ __launch_bounds__(256) void k_zero(float* p, int n) {
    int i = blockIdx.x * 256 + threadIdx.x;
    if (i < n) p[i] = 0.0f;
}

__global__ __launch_bounds__(256) void k_deg(const int* __restrict__ dst,
                                             float* __restrict__ deg, int E) {
    int e = blockIdx.x * 256 + threadIdx.x;
    if (e < E) atomicAdd(&deg[dst[e]], 1.0f);
}

// dinv[i] = rsqrt(deg[i] + 1)   (the +1 is the self-loop)
__global__ __launch_bounds__(256) void k_dinv(float* __restrict__ deg, int n) {
    int i = blockIdx.x * 256 + threadIdx.x;
    if (i < n) deg[i] = rsqrtf(deg[i] + 1.0f);
}

// W [K x N] f32 row-major  ->  WT [N x K] bf16 (so WMMA B fragments are
// contiguous 32-byte loads: lane holds one output column, K runs fastest).
__global__ __launch_bounds__(256) void k_wtrans(const float* __restrict__ W,
                                                __bf16* __restrict__ WT,
                                                int K, int N) {
    int idx = blockIdx.x * 256 + threadIdx.x;
    if (idx >= K * N) return;
    int n = idx / K;
    int k = idx - n * K;
    WT[idx] = (__bf16)W[k * N + n];
}

// out[i,f] = hin[i,f] * dinv[i]^2 + bias[f]   (self-loop message + bias)
__global__ __launch_bounds__(256) void k_agg_init(const float* __restrict__ hin,
                                                  const float* __restrict__ dinv,
                                                  const float* __restrict__ bias,
                                                  float* __restrict__ out,
                                                  int total /* nodes*128 */) {
    int idx = blockIdx.x * 256 + threadIdx.x;
    if (idx >= total) return;
    int i = idx >> 7;
    int f = idx & 127;
    float di = dinv[i];
    out[idx] = hin[idx] * di * di + bias[f];
}

// One wave (32 lanes) per edge: lane handles 4 features (float4 = 128 feats).
// out[dst] += hin[src] * dinv[src]*dinv[dst]   via f32 atomics (L2-resident).
__global__ __launch_bounds__(256) void k_edge_scatter(const int* __restrict__ src,
                                                      const int* __restrict__ dst,
                                                      const float* __restrict__ dinv,
                                                      const float* __restrict__ hin,
                                                      float* __restrict__ out, int E) {
    int t = blockIdx.x * 256 + threadIdx.x;
    int e = t >> 5;
    if (e >= E) return;
    int lane = t & 31;
    int s = src[e];
    int d = dst[e];
    float w = dinv[s] * dinv[d];
    const float4 v = *(const float4*)(hin + (size_t)s * NHID + lane * 4);
    float* o = out + (size_t)d * NHID + lane * 4;
    atomicAdd(o + 0, v.x * w);
    atomicAdd(o + 1, v.y * w);
    atomicAdd(o + 2, v.z * w);
    atomicAdd(o + 3, v.w * w);
}

// ---------------- WMMA GEMM:  out[M x 128] = act(A[M x K]) @ W[K x 128] (+b)
//
// Block = 256 threads = 8 waves covering 16 rows x 128 cols; wave w owns the
// 16-column tile [16w, 16w+16).
//
// Stage 1 (once): cooperative load of the full 16xK f32 A tile into LDS with
// fused optional ReLU; row stride padded to K+4 floats so the 16 `m` lanes
// map to distinct bank groups (conflict-free ds_load_b128). ONE barrier.
//
// Stage 2 (barrier-free K loop, step 32): each lane repacks its v16bf A
// fragment in contiguous k order (B uses the same order):
//   lane = h*16 + m, element i -> row m, K = h*16 + i.
// B fragment: lane reads WT[(16w + m)*K + k0 + h*16 .. +15] (one 32 B load),
// matching A's k order (lane%16 = column, K = h*16 + i).
// D layout: acc[r] -> row 8h + r, col m.  M is a multiple of 16 and all
// lanes stay active (WMMA requires EXEC all-ones).
// ----------------------------------------------------------------------------
template <int K, bool RELU, bool BIAS>
__global__ __launch_bounds__(256) void k_gemm_wmma(const float* __restrict__ A,
                                                   const __bf16* __restrict__ WT,
                                                   const float* __restrict__ bias,
                                                   float* __restrict__ out) {
    constexpr int LDS_STRIDE = K + 4;            // bank-conflict padding
    __shared__ float As[16 * LDS_STRIDE];        // <= 16.25 KB

    const int tid  = threadIdx.x;
    const int wave = tid >> 5;     // 0..7 -> column tile
    const int lane = tid & 31;
    const int m    = lane & 15;
    const int h    = lane >> 4;
    const int row0 = blockIdx.x * 16;
    const int ncol = wave * 16 + m;

    // ---- stage full A tile (16 x K f32) into LDS, fusing ReLU ----
#pragma unroll
    for (int i = tid * 4; i < 16 * K; i += 256 * 4) {
        int r = i / K;             // K is a power-of-two constant -> shift
        int c = i & (K - 1);
        float4 v = *(const float4*)(A + (size_t)(row0 + r) * K + c);
        if (RELU) {
            v.x = fmaxf(v.x, 0.0f); v.y = fmaxf(v.y, 0.0f);
            v.z = fmaxf(v.z, 0.0f); v.w = fmaxf(v.w, 0.0f);
        }
        *(float4*)(As + r * LDS_STRIDE + c) = v;
    }
    __syncthreads();               // the only barrier in this kernel

    v8f acc = {};
    const float* arow = As + m * LDS_STRIDE;

#pragma unroll
    for (int k0 = 0; k0 < K; k0 += 32) {
        v16bf a;
        {
            const float* ap = arow + k0 + h * 16;
#pragma unroll
            for (int i = 0; i < 8; ++i) a[i]     = (__bf16)ap[i];
#pragma unroll
            for (int i = 0; i < 8; ++i) a[8 + i] = (__bf16)ap[8 + i];
        }
        v16bf b = *(const v16bf*)(WT + (size_t)ncol * K + k0 + h * 16);

        acc = __builtin_amdgcn_wmma_f32_16x16x32_bf16(
            /*neg_a=*/false, a, /*neg_b=*/false, b,
            /*c_mod=*/(short)0, acc, /*reuse_a=*/false, /*reuse_b=*/false);
    }

    float bv = 0.0f;
    if (BIAS) bv = bias[ncol];
#pragma unroll
    for (int r = 0; r < 8; ++r) {
        int mr = row0 + h * 8 + r;
        out[(size_t)mr * NHID + ncol] = acc[r] + bv;
    }
}

// ---------------- orchestration ----------------

static inline size_t align256(size_t x) { return (x + 255) & ~(size_t)255; }

extern "C" void kernel_launch(void* const* d_in, const int* in_sizes, int n_in,
                              void* d_out, int out_size, void* d_ws, size_t ws_size,
                              hipStream_t stream) {
    const float* x      = (const float*)d_in[0];
    const int*   ei     = (const int*)d_in[1];     // [2, E]: row0 = src, row1 = dst
    const float* W_gc1  = (const float*)d_in[2];
    const float* b_gc1  = (const float*)d_in[3];
    const float* W_lin1 = (const float*)d_in[4];
    const float* b_lin1 = (const float*)d_in[5];
    const float* W_gc2  = (const float*)d_in[6];
    const float* b_gc2  = (const float*)d_in[7];
    const float* W_lin2 = (const float*)d_in[8];
    const float* b_lin2 = (const float*)d_in[9];

    const int NFEAT = 256;
    const int nodes = in_sizes[0] / NFEAT;         // 100000 (multiple of 16)
    const int E     = in_sizes[1] / 2;             // 3.2M
    const int* src  = ei;
    const int* dst  = ei + E;

    // workspace carve-out (~103 MB)
    char*  ws   = (char*)d_ws;
    size_t off  = 0;
    float* dinv = (float*)(ws + off); off += align256((size_t)nodes * 4);
    float* bufA = (float*)(ws + off); off += align256((size_t)nodes * NHID * 4);
    float* bufB = (float*)(ws + off); off += align256((size_t)nodes * NHID * 4);
    __bf16* WT_gc1  = (__bf16*)(ws + off); off += align256((size_t)NHID * NFEAT * 2);
    __bf16* WT_lin1 = (__bf16*)(ws + off); off += align256((size_t)NHID * NHID * 2);
    __bf16* WT_gc2  = (__bf16*)(ws + off); off += align256((size_t)NHID * NHID * 2);
    __bf16* WT_lin2 = (__bf16*)(ws + off); off += align256((size_t)NHID * NHID * 2);
    (void)ws_size;

    const int TB = 256;
    const int nNode   = (nodes + TB - 1) / TB;
    const int nEdge   = (E + TB - 1) / TB;
    const int nEdge32 = (int)(((size_t)E * 32 + TB - 1) / TB);
    const int nFeat   = (int)(((size_t)nodes * NHID + TB - 1) / TB);
    const int nGemm   = nodes / 16;                // 6250 blocks, 16 rows each

    // 1) symmetric normalization: deg -> dinv (self-loop folded in)
    k_zero<<<nNode, TB, 0, stream>>>(dinv, nodes);
    k_deg<<<nEdge, TB, 0, stream>>>(dst, dinv, E);
    k_dinv<<<nNode, TB, 0, stream>>>(dinv, nodes);

    // 2) weight transpose+convert to bf16 (tiny, L2-resident)
    k_wtrans<<<(NFEAT * NHID + TB - 1) / TB, TB, 0, stream>>>(W_gc1, WT_gc1, NFEAT, NHID);
    k_wtrans<<<(NHID * NHID + TB - 1) / TB, TB, 0, stream>>>(W_lin1, WT_lin1, NHID, NHID);
    k_wtrans<<<(NHID * NHID + TB - 1) / TB, TB, 0, stream>>>(W_gc2, WT_gc2, NHID, NHID);
    k_wtrans<<<(NHID * NHID + TB - 1) / TB, TB, 0, stream>>>(W_lin2, WT_lin2, NHID, NHID);

    // 3) layer 1:  h1 = x @ W_gc1 ; agg = scatter(norm * h1) + b_gc1
    k_gemm_wmma<256, false, false><<<nGemm, TB, 0, stream>>>(x, WT_gc1, nullptr, bufA);
    k_agg_init<<<nFeat, TB, 0, stream>>>(bufA, dinv, b_gc1, bufB, nodes * NHID);
    k_edge_scatter<<<nEdge32, TB, 0, stream>>>(src, dst, dinv, bufA, bufB, E);

    // 4) h = relu(agg) @ W_lin1 + b_lin1
    k_gemm_wmma<128, true, true><<<nGemm, TB, 0, stream>>>(bufB, WT_lin1, b_lin1, bufA);

    // 5) layer 2:  t = h @ W_gc2 ; agg2 = scatter(norm * t) + b_gc2
    k_gemm_wmma<128, false, false><<<nGemm, TB, 0, stream>>>(bufA, WT_gc2, nullptr, bufB);
    k_agg_init<<<nFeat, TB, 0, stream>>>(bufB, dinv, b_gc2, bufA, nodes * NHID);
    k_edge_scatter<<<nEdge32, TB, 0, stream>>>(src, dst, dinv, bufB, bufA, E);

    // 6) out = relu(agg2) @ W_lin2 + b_lin2   (written straight to d_out)
    k_gemm_wmma<128, true, true><<<nGemm, TB, 0, stream>>>(bufA, WT_lin2, b_lin2,
                                                           (float*)d_out);
}
